// SimpleCorrectionSeq2seq_63582695850518
// MI455X (gfx1250) — matrix operand order repaired and checked by reading, not compiled, not run on hardware
//
#include <hip/hip_runtime.h>

// ---------------------------------------------------------------------------
// Seq2seq GRU encoder/decoder with attention, MI455X (gfx1250, wave32).
// Dense matmuls: LDS double-buffered bf16 WMMA GEMM (v_wmma_f32_16x16x32_bf16),
// 8-wave blocks computing 128x128 C macro-tiles, 32x64 per wave (2x4 frags),
// one barrier per K-step, global prefetch 2 tiles ahead.
// ---------------------------------------------------------------------------

typedef __bf16 bf16;
typedef __attribute__((ext_vector_type(16))) __bf16    v16bf;
typedef __attribute__((ext_vector_type(8)))  float     v8f;
typedef __attribute__((ext_vector_type(4)))  unsigned  u32x4;

#define B_   4096
#define H_   512
#define L_   41
#define V_   128
#define TH_  1536   // 3*H
#define K2_  1024   // 2*H
#define BOS_ 2

// GEMM tiling
#define BM   128            // block macro-tile M
#define BN   128            // block macro-tile N
#define BK   32             // K step (one WMMA depth)
#define MT   2              // wave sub-tiles in M (32 rows)
#define NT   4              // wave sub-tiles in N (64 cols)
#define LDSP (BK + 8)       // padded LDS row stride (80B: bank-conflict-free)

// ---------------------------------------------------------------------------
//  C[M,ldc] = A[M,K] * W[N,K]^T + bias[N]   (A,W bf16; C f32 and/or bf16)
//  Requirements: M % 128 == 0, K % 32 == 0. N may be ragged (attn N=41).
// ---------------------------------------------------------------------------
__global__ __launch_bounds__(256) void gemm_bf16_wmma(
    const bf16* __restrict__ A, const bf16* __restrict__ W,
    const float* __restrict__ bias,
    float* __restrict__ C, bf16* __restrict__ Cbf,
    int M, int N, int K, int ldc, int relu_flag)
{
    __shared__ bf16 As[2][BM][LDSP];   // ping-pong buffers (40KB total of 320KB/WGP)
    __shared__ bf16 Bs[2][BN][LDSP];

    int nbn = (N + BN - 1) / BN;
    int m0  = (blockIdx.x / nbn) * BM;
    int n0  = (blockIdx.x % nbn) * BN;

    int tid  = threadIdx.x;
    int wave = tid >> 5;
    int lane = tid & 31;
    int wmb  = (wave >> 1) * (16 * MT);  // wave tile origin within block (M)
    int wnb  = (wave & 1)  * (16 * NT);  // wave tile origin within block (N)

    int half = lane >> 4;                // 16-lane half
    int l16  = lane & 15;

    // cooperative tile loaders: each thread moves 16 bf16 of A and of B
    int ldrow = tid >> 1;                // 0..127
    int ldcol = (tid & 1) * 16;          // 0 or 16
    int browg = n0 + ldrow;  if (browg >= N) browg = N - 1;   // clamp (no branchy loads)

    const bf16* Ald = A + (size_t)(m0 + ldrow) * K + ldcol;
    const bf16* Bld = W + (size_t)browg      * K + ldcol;

    union Frag { v16bf v; u32x4 q[2]; };
    v8f acc[MT][NT] = {};

    // ---- prologue: stage tile 0 into buffer 0 ----
    {
        u32x4 a0 = *(const u32x4*)(Ald);
        u32x4 a1 = *(const u32x4*)(Ald + 8);
        u32x4 b0 = *(const u32x4*)(Bld);
        u32x4 b1 = *(const u32x4*)(Bld + 8);
        *(u32x4*)(&As[0][ldrow][ldcol])     = a0;
        *(u32x4*)(&As[0][ldrow][ldcol + 8]) = a1;
        *(u32x4*)(&Bs[0][ldrow][ldcol])     = b0;
        *(u32x4*)(&Bs[0][ldrow][ldcol + 8]) = b1;
    }
    __syncthreads();

    int nk = K / BK;
    for (int ki = 0; ki < nk; ++ki) {
        int buf  = ki & 1;
        bool more = (ki + 1) < nk;

        // prefetch 2 tiles ahead into L2/WGP$ (global_prefetch_b8)
        if (ki + 2 < nk) {
            __builtin_prefetch(Ald + (ki + 2) * BK, 0, 1);
            __builtin_prefetch(Bld + (ki + 2) * BK, 0, 1);
        }

        // fetch next tile from global while this tile computes
        u32x4 a0, a1, b0, b1;
        if (more) {
            int k0 = (ki + 1) * BK;
            a0 = *(const u32x4*)(Ald + k0);
            a1 = *(const u32x4*)(Ald + k0 + 8);
            b0 = *(const u32x4*)(Bld + k0);
            b1 = *(const u32x4*)(Bld + k0 + 8);
        }

        // per-wave fragments from LDS[buf] (documented CDNA5 bf16 layouts)
        Frag a[MT], b[NT];
#pragma unroll
        for (int i = 0; i < MT; ++i) {
            const bf16* r = &As[buf][wmb + i * 16 + l16][0];
            a[i].q[0] = *(const u32x4*)(r + 8 * half);        // K 0..7 / 8..15
            a[i].q[1] = *(const u32x4*)(r + 16 + 8 * half);   // K 16..23 / 24..31
        }
#pragma unroll
        for (int j = 0; j < NT; ++j) {
            const bf16* r = &Bs[buf][wnb + j * 16 + l16][0];
            b[j].q[0] = *(const u32x4*)(r + 8 * half);        // K 0..7 / 8..15
            b[j].q[1] = *(const u32x4*)(r + 16 + 8 * half);   // K 16..23 / 24..31
        }
#pragma unroll
        for (int i = 0; i < MT; ++i)
#pragma unroll
            for (int j = 0; j < NT; ++j)
                acc[i][j] = __builtin_amdgcn_wmma_f32_16x16x32_bf16(
                    false, a[i].v, false, b[j].v, (short)0, acc[i][j], false, false);

        // stage next tile into the other buffer; single barrier per iteration
        if (more) {
            int nb = buf ^ 1;
            *(u32x4*)(&As[nb][ldrow][ldcol])     = a0;
            *(u32x4*)(&As[nb][ldrow][ldcol + 8]) = a1;
            *(u32x4*)(&Bs[nb][ldrow][ldcol])     = b0;
            *(u32x4*)(&Bs[nb][ldrow][ldcol + 8]) = b1;
        }
        __syncthreads();
    }

    // store: C layout per tile: VGPR r -> row r+8*half, lane l16 -> column
#pragma unroll
    for (int j = 0; j < NT; ++j) {
        int ncol = n0 + wnb + j * 16 + l16;
        if (ncol >= N) continue;
        float bv = bias ? bias[ncol] : 0.0f;
#pragma unroll
        for (int i = 0; i < MT; ++i) {
#pragma unroll
            for (int r = 0; r < 8; ++r) {
                float x = acc[i][j][r] + bv;
                if (relu_flag) x = fmaxf(x, 0.0f);
                int mrow = m0 + wmb + i * 16 + r + 8 * half;
                size_t idx = (size_t)mrow * ldc + ncol;
                if (C)   C[idx]   = x;
                if (Cbf) Cbf[idx] = (bf16)x;
            }
        }
    }
}

// --------------------------- elementwise kernels ---------------------------

__global__ void cvt_f32_bf16(const float* __restrict__ s, bf16* __restrict__ d, int n) {
    int i = blockIdx.x * blockDim.x + threadIdx.x;
    if (i < n) d[i] = (bf16)s[i];
}

__global__ void zero_f32(float* p, int n) {
    int i = blockIdx.x * blockDim.x + threadIdx.x;
    if (i < n) p[i] = 0.0f;
}

// dst[b*dstride + doff + j] = emb[tok[b]][j]   (tok==null -> BOS)
__global__ void embed_gather(const int* __restrict__ tok, const bf16* __restrict__ emb,
                             bf16* __restrict__ dst, int dstride, int doff) {
    int i = blockIdx.x * blockDim.x + threadIdx.x;
    if (i >= B_ * H_) return;
    int b = i / H_, j = i % H_;
    int t = tok ? tok[b] : BOS_;
    dst[(size_t)b * dstride + doff + j] = emb[(size_t)t * H_ + j];
}

// strided bf16 row copy: dst[b*ds + doff + j] = src[b*ss + j]
__global__ void copy_rows_bf16(const bf16* __restrict__ src, int ss,
                               bf16* __restrict__ dst, int ds, int doff) {
    int i = blockIdx.x * blockDim.x + threadIdx.x;
    if (i >= B_ * H_) return;
    int b = i / H_, j = i % H_;
    dst[(size_t)b * ds + doff + j] = src[(size_t)b * ss + j];
}

// GRU gate combine (PyTorch order r,z,n). Writes f32 h and bf16 mirror.
__global__ void gru_combine(const float* __restrict__ gi, const float* __restrict__ gh,
                            const float* __restrict__ h,
                            float* __restrict__ ho, bf16* __restrict__ hob) {
    int i = blockIdx.x * blockDim.x + threadIdx.x;
    if (i >= B_ * H_) return;
    int b = i / H_, j = i % H_;
    size_t g = (size_t)b * TH_ + j;
    float ir = gi[g], iz = gi[g + H_], in = gi[g + 2 * H_];
    float hr = gh[g], hz = gh[g + H_], hn = gh[g + 2 * H_];
    float r = 1.0f / (1.0f + __expf(-(ir + hr)));
    float z = 1.0f / (1.0f + __expf(-(iz + hz)));
    float n = tanhf(in + r * hn);
    float hv = (1.0f - z) * n + z * h[i];
    ho[i] = hv;
    hob[i] = (bf16)hv;
}

// enc_pad[t][:] = h[0][:]
__global__ void copy_row0(const float* __restrict__ h, float* __restrict__ dst) {
    int j = blockIdx.x * blockDim.x + threadIdx.x;
    if (j < H_) dst[j] = h[j];
}

// encoder_outputs[b][l][j] = enc_pad[l][j]
__global__ void bcast_enc(const float* __restrict__ enc_pad, float* __restrict__ out) {
    size_t i = (size_t)blockIdx.x * blockDim.x + threadIdx.x;
    size_t total = (size_t)B_ * L_ * H_;
    if (i >= total) return;
    out[i] = enc_pad[i % ((size_t)L_ * H_)];
}

// in-place softmax over L_ entries of a row with stride ld
__global__ void softmax_rows(float* __restrict__ x, int ld) {
    int b = blockIdx.x * blockDim.x + threadIdx.x;
    if (b >= B_) return;
    float* row = x + (size_t)b * ld;
    float m = row[0];
    for (int l = 1; l < L_; ++l) m = fmaxf(m, row[l]);
    float s = 0.0f;
    for (int l = 0; l < L_; ++l) { float e = __expf(row[l] - m); row[l] = e; s += e; }
    float inv = 1.0f / s;
    for (int l = 0; l < L_; ++l) row[l] *= inv;
}

// ctx[b][j] = sum_l aw[b][l] * enc_pad[l][j]   -> bf16 into cat2 right half
__global__ void ctx_kernel(const float* __restrict__ aw, int lda,
                           const float* __restrict__ enc_pad,
                           bf16* __restrict__ dst, int ds, int doff) {
    int i = blockIdx.x * blockDim.x + threadIdx.x;
    if (i >= B_ * H_) return;
    int b = i / H_, j = i % H_;
    float s = 0.0f;
    const float* a = aw + (size_t)b * lda;
#pragma unroll 1
    for (int l = 0; l < L_; ++l) s += a[l] * enc_pad[(size_t)l * H_ + j];
    dst[(size_t)b * ds + doff + j] = (bf16)s;
}

// scores[b] += (tgt[b] != PAD) * (log_softmax(logits[b])[tgt[b]])
__global__ void nll_acc(const float* __restrict__ logits, int ld,
                        const int* __restrict__ tgt, float* __restrict__ scores) {
    int b = blockIdx.x * blockDim.x + threadIdx.x;
    if (b >= B_) return;
    const float* row = logits + (size_t)b * ld;
    float m = row[0];
    for (int v = 1; v < V_; ++v) m = fmaxf(m, row[v]);
    float s = 0.0f;
    for (int v = 0; v < V_; ++v) s += __expf(row[v] - m);
    float logZ = m + __logf(s);
    int t = tgt[b];
    if (t != 0) scores[b] += row[t] - logZ;
}

// ------------------------------- host side ---------------------------------

static inline void launch_gemm(const bf16* A, const bf16* W, const float* bias,
                               float* C, bf16* Cbf, int M, int N, int K, int ldc,
                               int relu, hipStream_t s) {
    int blocks = (M / BM) * ((N + BN - 1) / BN);
    gemm_bf16_wmma<<<blocks, 256, 0, s>>>(A, W, bias, C, Cbf, M, N, K, ldc, relu);
}

#define EW(n) dim3(((n) + 255) / 256), dim3(256)

extern "C" void kernel_launch(void* const* d_in, const int* in_sizes, int n_in,
                              void* d_out, int out_size, void* d_ws, size_t ws_size,
                              hipStream_t stream) {
    const int*   input      = (const int*)d_in[0];     // [L, B]
    const float* enc_hidden = (const float*)d_in[1];   // [1, B, H]
    const int*   target     = (const int*)d_in[2];     // [L, B]
    const float* enc_emb    = (const float*)d_in[3];
    const float* enc_Wih    = (const float*)d_in[4];
    const float* enc_Whh    = (const float*)d_in[5];
    const float* enc_bih    = (const float*)d_in[6];
    const float* enc_bhh    = (const float*)d_in[7];
    const float* dec_emb    = (const float*)d_in[8];
    const float* attn_W     = (const float*)d_in[9];
    const float* attn_b     = (const float*)d_in[10];
    const float* comb_W     = (const float*)d_in[11];
    const float* comb_b     = (const float*)d_in[12];
    const float* dec_Wih    = (const float*)d_in[13];
    const float* dec_Whh    = (const float*)d_in[14];
    const float* dec_bih    = (const float*)d_in[15];
    const float* dec_bhh    = (const float*)d_in[16];
    const float* out_W      = (const float*)d_in[17];
    const float* out_b      = (const float*)d_in[18];

    float* scores  = (float*)d_out;          // [B]
    float* enc_out = scores + B_;            // [B, L, H]

    // ---- workspace bump allocator ----
    char* ws = (char*)d_ws; size_t off = 0;
    auto alloc = [&](size_t bytes) -> void* {
        void* p = ws + off; off = (off + bytes + 255) & ~(size_t)255; return p;
    };
    bf16* w_enc_emb = (bf16*)alloc((size_t)V_ * H_ * 2);
    bf16* w_enc_Wih = (bf16*)alloc((size_t)TH_ * H_ * 2);
    bf16* w_enc_Whh = (bf16*)alloc((size_t)TH_ * H_ * 2);
    bf16* w_dec_emb = (bf16*)alloc((size_t)V_ * H_ * 2);
    bf16* w_attn_W  = (bf16*)alloc((size_t)L_ * K2_ * 2);
    bf16* w_comb_W  = (bf16*)alloc((size_t)H_ * K2_ * 2);
    bf16* w_dec_Wih = (bf16*)alloc((size_t)TH_ * H_ * 2);
    bf16* w_dec_Whh = (bf16*)alloc((size_t)TH_ * H_ * 2);
    bf16* w_out_W   = (bf16*)alloc((size_t)V_ * H_ * 2);

    float* h_f32   = (float*)alloc((size_t)B_ * H_ * 4);
    bf16*  h_bf    = (bf16*) alloc((size_t)B_ * H_ * 2);
    bf16*  x_bf    = (bf16*) alloc((size_t)B_ * H_ * 2);     // encoder embedding
    float* gi      = (float*)alloc((size_t)B_ * TH_ * 4);
    float* gh      = (float*)alloc((size_t)B_ * TH_ * 4);
    float* enc_pad = (float*)alloc((size_t)L_ * H_ * 4);
    bf16*  cat1    = (bf16*) alloc((size_t)B_ * K2_ * 2);    // [emb | h]  for attn
    float* attnlog = (float*)alloc((size_t)B_ * 48 * 4);     // ld=48, 41 valid
    bf16*  cat2    = (bf16*) alloc((size_t)B_ * K2_ * 2);    // [emb | ctx] for combine
    bf16*  xd_bf   = (bf16*) alloc((size_t)B_ * H_ * 2);     // combine output
    float* logits  = (float*)alloc((size_t)B_ * V_ * 4);

    // ---- one-time per call: weight conversion, state init ----
    cvt_f32_bf16<<<EW(V_*H_),   0, stream>>>(enc_emb, w_enc_emb, V_*H_);
    cvt_f32_bf16<<<EW(TH_*H_),  0, stream>>>(enc_Wih, w_enc_Wih, TH_*H_);
    cvt_f32_bf16<<<EW(TH_*H_),  0, stream>>>(enc_Whh, w_enc_Whh, TH_*H_);
    cvt_f32_bf16<<<EW(V_*H_),   0, stream>>>(dec_emb, w_dec_emb, V_*H_);
    cvt_f32_bf16<<<EW(L_*K2_),  0, stream>>>(attn_W,  w_attn_W,  L_*K2_);
    cvt_f32_bf16<<<EW(H_*K2_),  0, stream>>>(comb_W,  w_comb_W,  H_*K2_);
    cvt_f32_bf16<<<EW(TH_*H_),  0, stream>>>(dec_Wih, w_dec_Wih, TH_*H_);
    cvt_f32_bf16<<<EW(TH_*H_),  0, stream>>>(dec_Whh, w_dec_Whh, TH_*H_);
    cvt_f32_bf16<<<EW(V_*H_),   0, stream>>>(out_W,   w_out_W,   V_*H_);

    hipMemcpyAsync(h_f32, enc_hidden, (size_t)B_*H_*4, hipMemcpyDeviceToDevice, stream);
    cvt_f32_bf16<<<EW(B_*H_), 0, stream>>>(enc_hidden, h_bf, B_*H_);
    zero_f32<<<EW(B_), 0, stream>>>(scores, B_);

    // ---- encoder: 41 GRU steps ----
    for (int t = 0; t < L_; ++t) {
        embed_gather<<<EW(B_*H_), 0, stream>>>(input + (size_t)t*B_, w_enc_emb, x_bf, H_, 0);
        launch_gemm(x_bf, w_enc_Wih, enc_bih, gi, nullptr, B_, TH_, H_, TH_, 0, stream);
        launch_gemm(h_bf, w_enc_Whh, enc_bhh, gh, nullptr, B_, TH_, H_, TH_, 0, stream);
        gru_combine<<<EW(B_*H_), 0, stream>>>(gi, gh, h_f32, h_f32, h_bf);
        copy_row0<<<EW(H_), 0, stream>>>(h_f32, enc_pad + (size_t)t*H_);
    }

    // encoder_outputs = broadcast(enc_pad) -> d_out[B_ ..]
    {
        size_t total = (size_t)B_ * L_ * H_;
        bcast_enc<<<dim3((unsigned)((total + 255) / 256)), dim3(256), 0, stream>>>(enc_pad, enc_out);
    }

    // ---- decoder: 41 steps, teacher forcing ----
    for (int t = 0; t < L_; ++t) {
        const int* tok = (t == 0) ? nullptr : target + (size_t)(t-1)*B_;
        // cat1 = [emb | h_prev], cat2 left = emb
        embed_gather<<<EW(B_*H_), 0, stream>>>(tok, w_dec_emb, cat1, K2_, 0);
        embed_gather<<<EW(B_*H_), 0, stream>>>(tok, w_dec_emb, cat2, K2_, 0);
        copy_rows_bf16<<<EW(B_*H_), 0, stream>>>(h_bf, H_, cat1, K2_, H_);
        // attention weights
        launch_gemm(cat1, w_attn_W, attn_b, attnlog, nullptr, B_, L_, K2_, 48, 0, stream);
        softmax_rows<<<EW(B_), 0, stream>>>(attnlog, 48);
        // context -> cat2 right half
        ctx_kernel<<<EW(B_*H_), 0, stream>>>(attnlog, 48, enc_pad, cat2, K2_, H_);
        // x = relu(cat2 @ comb_W^T + comb_b)
        launch_gemm(cat2, w_comb_W, comb_b, nullptr, xd_bf, B_, H_, K2_, H_, 1, stream);
        // GRU
        launch_gemm(xd_bf, w_dec_Wih, dec_bih, gi, nullptr, B_, TH_, H_, TH_, 0, stream);
        launch_gemm(h_bf,  w_dec_Whh, dec_bhh, gh, nullptr, B_, TH_, H_, TH_, 0, stream);
        gru_combine<<<EW(B_*H_), 0, stream>>>(gi, gh, h_f32, h_f32, h_bf);
        // output logits + NLL accumulation
        launch_gemm(h_bf, w_out_W, out_b, logits, nullptr, B_, V_, H_, V_, 0, stream);
        nll_acc<<<EW(B_), 0, stream>>>(logits, V_, target + (size_t)t*B_, scores);
    }
}
